// CausalSelfAttention_26113401160414
// MI455X (gfx1250) — matrix operand order repaired and checked by reading, not compiled, not run on hardware
//
#include <hip/hip_runtime.h>
#include <hip/hip_bf16.h>

#define D_DIM 1024
#define B_DIM 4
#define N_DIM 2048
#define M_ROWS (B_DIM * N_DIM)   // 8192

typedef __attribute__((ext_vector_type(16))) __bf16 v16bf;
typedef __attribute__((ext_vector_type(8)))  float  v8f;
typedef __attribute__((ext_vector_type(8)))  __bf16 bf16x8;
typedef __attribute__((ext_vector_type(4)))  unsigned int v4u;
typedef __attribute__((ext_vector_type(8)))  int v8i;
typedef __attribute__((ext_vector_type(4)))  int v4i;

union V16U { v16bf v; bf16x8 h[2]; };

#define WMMA_BF16(a, b, c) \
    __builtin_amdgcn_wmma_f32_16x16x32_bf16(false, (a), false, (b), (short)0, (c), false, false)

__device__ __forceinline__ void split_bf16(float f, __bf16& h, __bf16& l) {
    h = (__bf16)f;
    l = (__bf16)(f - (float)h);
}

// A operand (16x32 bf16). Lanes 0-15 hold K 0..15, lanes 16-31 K 16..31 (matches B).
__device__ __forceinline__ v16bf load_a_tile(const __bf16* base, int row, int kc, int lane) {
    const __bf16* p = base + (size_t)row * D_DIM + kc + ((lane >> 4) << 4);
    V16U u;
    u.h[0] = *(const bf16x8*)p;
    u.h[1] = *(const bf16x8*)(p + 8);
    return u.v;
}

// B operand (32x16 bf16) from row-major [col][K]. Same K order as A: lanes 0-15 hold K 0..15, lanes 16-31 K 16..31.
__device__ __forceinline__ v16bf load_b_tile(const __bf16* base, int row, int kc, int lane) {
    const __bf16* p = base + (size_t)row * D_DIM + kc + ((lane >> 4) << 4);
    V16U u;
    u.h[0] = *(const bf16x8*)p;
    u.h[1] = *(const bf16x8*)(p + 8);
    return u.v;
}

// B operand from an LDS-resident 16x1024 panel (row-major).
__device__ __forceinline__ v16bf load_b_lds(const __bf16* panel, int row, int kc, int lane) {
    const __bf16* p = panel + row * D_DIM + kc + ((lane >> 4) << 4);
    V16U u;
    u.h[0] = *(const bf16x8*)p;
    u.h[1] = *(const bf16x8*)(p + 8);
    return u.v;
}

// Tensor Data Mover: DMA a 16-row x 1024-elem bf16 panel (global, row-major) into LDS.
// D# per CDNA5 ISA ch.8: group0 = {count, lds_addr, global_addr, type=2},
// group1 = {data_size=2B, tensor_dim0/1, tile_dim0/1, tensor_dim0_stride}.
__device__ __forceinline__ void tdm_load_panel(const __bf16* gptr, const __bf16* lptr) {
    unsigned long long ga = (unsigned long long)gptr;
    unsigned la = (unsigned)(unsigned long long)lptr;   // LDS aperture: addr[31:0] == LDS byte offset
    v4u g0;
    g0[0] = 1u;                                          // count = 1 valid descriptor
    g0[1] = la;                                          // lds_addr
    g0[2] = (unsigned)ga;                                // global_addr[31:0]
    g0[3] = (unsigned)((ga >> 32) & 0x1FFFFFFu) | 0x80000000u; // global_addr[56:32] | type=2<<30
    const unsigned dim0 = 1024u, dim1 = 16u, t0 = 1024u, t1 = 16u, stride = 1024u;
    v8i g1;
    g1[0] = 0x00010000;                                  // workgroup_mask=0, data_size=1 (2 bytes)
    g1[1] = (int)((dim0 & 0xFFFFu) << 16);               // tensor_dim0[15:0] -> bits 63:48
    g1[2] = (int)(((dim0 >> 16) & 0xFFFFu) | ((dim1 & 0xFFFFu) << 16)); // dim0 hi | dim1 lo
    g1[3] = (int)(((dim1 >> 16) & 0xFFFFu) | ((t0 & 0xFFFFu) << 16));   // dim1 hi | tile_dim0
    g1[4] = (int)(t1 & 0xFFFFu);                         // tile_dim1 (tile_dim2 = 0)
    g1[5] = (int)stride;                                 // tensor_dim0_stride[31:0]
    g1[6] = 0;                                           // stride hi | tensor_dim1_stride lo (unused, 2D)
    g1[7] = 0;
    v4i z4 = {0, 0, 0, 0};
    v8i z8 = {0, 0, 0, 0, 0, 0, 0, 0};
    __builtin_amdgcn_tensor_load_to_lds(g0, g1, z4, z4, z8, 0);
}

// ---------------------------------------------------------------- split fp32 -> bf16 hi/lo
__global__ __launch_bounds__(256)
void split_kernel(const float* __restrict__ src, __bf16* __restrict__ h,
                  __bf16* __restrict__ l, int n) {
    int i = blockIdx.x * 256 + threadIdx.x;
    if (i < n) {
        __bf16 hb, lb;
        split_bf16(src[i], hb, lb);
        h[i] = hb;
        l[i] = lb;
    }
}

// ---------------------------------------------------------------- fused projection y = x W^T + b
// 8 waves = 4 (M) x 2 (N); each wave computes a 32x32 output tile (4 accumulators).
__global__ __launch_bounds__(256)
void qkv_kernel(const __bf16* __restrict__ xh, const __bf16* __restrict__ xl,
                const __bf16* __restrict__ wh, const __bf16* __restrict__ wl,
                const float* __restrict__ bias,
                __bf16* __restrict__ oh, __bf16* __restrict__ ol,
                float* __restrict__ of) {
    const int tid  = threadIdx.x;
    const int lane = tid & 31;
    const int w    = tid >> 5;
    const int lr   = lane & 15;
    const int hv   = lane >> 4;
    const int m0   = blockIdx.x * 128 + (w & 3) * 32;
    const int n0   = blockIdx.y * 64 + (w >> 2) * 32;

    v8f c00 = {}, c01 = {}, c10 = {}, c11 = {};
    for (int kc = 0; kc < D_DIM; kc += 32) {
        v16bf a0h = load_a_tile(xh, m0 + lr, kc, lane);
        v16bf a0l = load_a_tile(xl, m0 + lr, kc, lane);
        v16bf a1h = load_a_tile(xh, m0 + 16 + lr, kc, lane);
        v16bf a1l = load_a_tile(xl, m0 + 16 + lr, kc, lane);
        v16bf b0h = load_b_tile(wh, n0 + lr, kc, lane);
        v16bf b0l = load_b_tile(wl, n0 + lr, kc, lane);
        v16bf b1h = load_b_tile(wh, n0 + 16 + lr, kc, lane);
        v16bf b1l = load_b_tile(wl, n0 + 16 + lr, kc, lane);
        c00 = WMMA_BF16(a0h, b0h, c00);
        c00 = WMMA_BF16(a0h, b0l, c00);
        c00 = WMMA_BF16(a0l, b0h, c00);
        c01 = WMMA_BF16(a0h, b1h, c01);
        c01 = WMMA_BF16(a0h, b1l, c01);
        c01 = WMMA_BF16(a0l, b1h, c01);
        c10 = WMMA_BF16(a1h, b0h, c10);
        c10 = WMMA_BF16(a1h, b0l, c10);
        c10 = WMMA_BF16(a1l, b0h, c10);
        c11 = WMMA_BF16(a1h, b1h, c11);
        c11 = WMMA_BF16(a1h, b1l, c11);
        c11 = WMMA_BF16(a1l, b1h, c11);
    }
    const float bias0 = bias[n0 + lr];
    const float bias1 = bias[n0 + 16 + lr];
#pragma unroll
    for (int r = 0; r < 8; r++) {
        const int M0 = m0 + r + 8 * hv;
        const int M1 = M0 + 16;
        float y00 = c00[r] + bias0, y01 = c01[r] + bias1;
        float y10 = c10[r] + bias0, y11 = c11[r] + bias1;
        if (of) {
            of[(size_t)M0 * D_DIM + n0 + lr]      = y00;
            of[(size_t)M0 * D_DIM + n0 + 16 + lr] = y01;
            of[(size_t)M1 * D_DIM + n0 + lr]      = y10;
            of[(size_t)M1 * D_DIM + n0 + 16 + lr] = y11;
        } else {
            __bf16 hb, lb;
            split_bf16(y00, hb, lb);
            oh[(size_t)M0 * D_DIM + n0 + lr] = hb;
            ol[(size_t)M0 * D_DIM + n0 + lr] = lb;
            split_bf16(y01, hb, lb);
            oh[(size_t)M0 * D_DIM + n0 + 16 + lr] = hb;
            ol[(size_t)M0 * D_DIM + n0 + 16 + lr] = lb;
            split_bf16(y10, hb, lb);
            oh[(size_t)M1 * D_DIM + n0 + lr] = hb;
            ol[(size_t)M1 * D_DIM + n0 + lr] = lb;
            split_bf16(y11, hb, lb);
            oh[(size_t)M1 * D_DIM + n0 + 16 + lr] = hb;
            ol[(size_t)M1 * D_DIM + n0 + 16 + lr] = lb;
        }
    }
}

// ---------------------------------------------------------------- upper-triangle score stats
// Workgroup = 128-row i-strip (wave w owns rows i0+16w..+15). All waves sweep the same
// j-tile; the 16x1024 K panel (hi+lo) is TDM-staged into double-buffered LDS.
__global__ __launch_bounds__(256)
void attn_stats_kernel(const __bf16* __restrict__ qh, const __bf16* __restrict__ ql,
                       const __bf16* __restrict__ kh, const __bf16* __restrict__ kl,
                       float* __restrict__ alpha, float* __restrict__ beta) {
    extern __shared__ __bf16 smem[];   // 4 panels of 16*1024 bf16 = 128 KB
    __bf16* pan[2][2];
    pan[0][0] = smem;
    pan[0][1] = smem + 16 * 1024;
    pan[1][0] = smem + 32 * 1024;
    pan[1][1] = smem + 48 * 1024;

    const int b    = blockIdx.y;
    const int i0   = blockIdx.x * 128;
    const int tid  = threadIdx.x;
    const int lane = tid & 31;
    const int w    = tid >> 5;
    const int lr   = lane & 15;
    const int hv   = lane >> 4;
    const int iw0  = i0 + w * 16;      // this wave's row base

    const __bf16* qhb = qh + (size_t)b * N_DIM * D_DIM;
    const __bf16* qlb = ql + (size_t)b * N_DIM * D_DIM;
    const __bf16* khb = kh + (size_t)b * N_DIM * D_DIM;
    const __bf16* klb = kl + (size_t)b * N_DIM * D_DIM;

    const float NEG_INF = -__builtin_inff();
    float m_run[8], l_run[8], sii[8];
#pragma unroll
    for (int r = 0; r < 8; r++) { m_run[r] = NEG_INF; l_run[r] = 0.f; sii[r] = 0.f; }

    const int jt0 = i0 / 16;
    int cur = 0;
    if (w == 0) {
        tdm_load_panel(khb + (size_t)jt0 * 16 * D_DIM, pan[0][0]);
        tdm_load_panel(klb + (size_t)jt0 * 16 * D_DIM, pan[0][1]);
    }

    for (int jt = jt0; jt < N_DIM / 16; jt++) {
        if (w == 0) __builtin_amdgcn_s_wait_tensorcnt(0);
        __syncthreads();                                  // panel[cur] visible to all waves
        if (w == 0 && jt + 1 < N_DIM / 16) {              // prefetch next panel
            tdm_load_panel(khb + (size_t)(jt + 1) * 16 * D_DIM, pan[cur ^ 1][0]);
            tdm_load_panel(klb + (size_t)(jt + 1) * 16 * D_DIM, pan[cur ^ 1][1]);
        }
        const int j0 = jt * 16;
        if (j0 + 15 >= iw0) {                             // tile intersects j >= i for this wave
            v8f c = {};
            for (int kc = 0; kc < D_DIM; kc += 32) {
                v16bf ah = load_a_tile(qhb, iw0 + lr, kc, lane);
                v16bf al = load_a_tile(qlb, iw0 + lr, kc, lane);
                v16bf bh = load_b_lds(pan[cur][0], lr, kc, lane);
                v16bf bl = load_b_lds(pan[cur][1], lr, kc, lane);
                c = WMMA_BF16(ah, bh, c);
                c = WMMA_BF16(ah, bl, c);
                c = WMMA_BF16(al, bh, c);
            }
            const bool isdiag = (j0 == iw0);
#pragma unroll
            for (int r = 0; r < 8; r++) {
                const int M = r + 8 * hv;
                float sc = c[r] * 0.03125f;               // 1/sqrt(1024)
                if (isdiag && lr == M) sii[r] = sc;       // diagonal s_ii
                const bool valid = (j0 + lr) >= (iw0 + M);
                float sm = valid ? sc : NEG_INF;
#pragma unroll
                for (int off = 1; off < 16; off <<= 1)
                    sm = fmaxf(sm, __shfl_xor(sm, off, 32));
                float e = valid ? __expf(sc - sm) : 0.f;
#pragma unroll
                for (int off = 1; off < 16; off <<= 1)
                    e += __shfl_xor(e, off, 32);
                if (sm > NEG_INF) {                       // active tiles always have >=1 valid/row
                    const float mo = m_run[r];
                    const float mn = fmaxf(mo, sm);
                    l_run[r] = l_run[r] * __expf(mo - mn) + e * __expf(sm - mn);
                    m_run[r] = mn;
                }
            }
        }
        __syncthreads();                                  // all done with panel[cur] before reuse
        cur ^= 1;
    }

    // Per-wave finalize: the diagonal lane of each row computes alpha/beta.
#pragma unroll
    for (int r = 0; r < 8; r++) {
        const int M = r + 8 * hv;
        if (lr == M) {
            const int   i  = iw0 + M;
            const float ms = m_run[r];
            const float mt = (i > 0) ? fmaxf(ms, 0.f) : ms;  // zeros from j<i join the softmax
            const float Z  = (float)i * __expf(-mt) + l_run[r] * __expf(ms - mt);
            alpha[b * N_DIM + i] = __expf(-mt) / Z;
            beta [b * N_DIM + i] = __expf(sii[r] - mt) / Z;
        }
    }
}

// ---------------------------------------------------------------- V prefix scan (blocked)
__global__ __launch_bounds__(256)
void vbsum_kernel(const float* __restrict__ v, float* __restrict__ vbs) {
    const int b   = blockIdx.x >> 5;
    const int blk = blockIdx.x & 31;
    const int d   = threadIdx.x * 4;
    const float4* vp = (const float4*)(v + ((size_t)(b * N_DIM + blk * 64)) * D_DIM + d);
    float4 s = {0.f, 0.f, 0.f, 0.f};
    for (int i = 0; i < 64; i++) {
        float4 t = vp[i * (D_DIM / 4)];
        s.x += t.x; s.y += t.y; s.z += t.z; s.w += t.w;
    }
    *(float4*)(vbs + ((size_t)(b * 32 + blk)) * D_DIM + d) = s;
}

__global__ __launch_bounds__(256)
void finalize_kernel(const float* __restrict__ v, const float* __restrict__ vbs,
                     const float* __restrict__ alpha, const float* __restrict__ beta,
                     float* __restrict__ out) {
    const int b   = blockIdx.x >> 5;
    const int blk = blockIdx.x & 31;
    const int d   = threadIdx.x * 4;
    float4 base = {0.f, 0.f, 0.f, 0.f};
    for (int p = 0; p < blk; p++) {
        float4 t = *(const float4*)(vbs + ((size_t)(b * 32 + p)) * D_DIM + d);
        base.x += t.x; base.y += t.y; base.z += t.z; base.w += t.w;
    }
    const float4* vp = (const float4*)(v + ((size_t)(b * N_DIM + blk * 64)) * D_DIM + d);
    float4* op = (float4*)(out + ((size_t)(b * N_DIM + blk * 64)) * D_DIM + d);
    const int ib = b * N_DIM + blk * 64;
    for (int i = 0; i < 64; i++) {
        const float a  = alpha[ib + i];
        const float be = beta [ib + i];
        float4 t = vp[i * (D_DIM / 4)];
        float4 o;
        o.x = a * base.x + be * t.x;
        o.y = a * base.y + be * t.y;
        o.z = a * base.z + be * t.z;
        o.w = a * base.w + be * t.w;
        op[i * (D_DIM / 4)] = o;
        base.x += t.x; base.y += t.y; base.z += t.z; base.w += t.w;
    }
}

// ----------------------------------------------------------------
extern "C" void kernel_launch(void* const* d_in, const int* in_sizes, int n_in,
                              void* d_out, int out_size, void* d_ws, size_t ws_size,
                              hipStream_t stream) {
    const float* x  = (const float*)d_in[0];
    const float* wq = (const float*)d_in[1];
    const float* bq = (const float*)d_in[2];
    const float* wk = (const float*)d_in[3];
    const float* bk = (const float*)d_in[4];
    const float* wv = (const float*)d_in[5];
    const float* bv = (const float*)d_in[6];

    char* ws = (char*)d_ws;
    size_t off = 0;
    auto carve = [&](size_t bytes) -> void* {
        void* p = ws + off;
        off += (bytes + 255) & ~(size_t)255;
        return p;
    };
    const size_t xe = (size_t)M_ROWS * D_DIM;
    const size_t we = (size_t)D_DIM * D_DIM;
    __bf16* xh  = (__bf16*)carve(xe * 2);
    __bf16* xl  = (__bf16*)carve(xe * 2);
    __bf16* wqh = (__bf16*)carve(we * 2);
    __bf16* wql = (__bf16*)carve(we * 2);
    __bf16* wkh = (__bf16*)carve(we * 2);
    __bf16* wkl = (__bf16*)carve(we * 2);
    __bf16* wvh = (__bf16*)carve(we * 2);
    __bf16* wvl = (__bf16*)carve(we * 2);
    __bf16* qh  = (__bf16*)carve(xe * 2);
    __bf16* qlo = (__bf16*)carve(xe * 2);
    __bf16* kh  = (__bf16*)carve(xe * 2);
    __bf16* klo = (__bf16*)carve(xe * 2);
    float*  vbuf  = (float*)carve(xe * 4);
    float*  vbs   = (float*)carve((size_t)B_DIM * 32 * D_DIM * 4);
    float*  alpha = (float*)carve((size_t)B_DIM * N_DIM * 4);
    float*  beta  = (float*)carve((size_t)B_DIM * N_DIM * 4);

    split_kernel<<<(int)((xe + 255) / 256), 256, 0, stream>>>(x, xh, xl, (int)xe);
    split_kernel<<<(int)((we + 255) / 256), 256, 0, stream>>>(wq, wqh, wql, (int)we);
    split_kernel<<<(int)((we + 255) / 256), 256, 0, stream>>>(wk, wkh, wkl, (int)we);
    split_kernel<<<(int)((we + 255) / 256), 256, 0, stream>>>(wv, wvh, wvl, (int)we);

    dim3 g1(M_ROWS / 128, D_DIM / 64);
    qkv_kernel<<<g1, 256, 0, stream>>>(xh, xl, wqh, wql, bq, qh, qlo, nullptr);
    qkv_kernel<<<g1, 256, 0, stream>>>(xh, xl, wkh, wkl, bk, kh, klo, nullptr);
    qkv_kernel<<<g1, 256, 0, stream>>>(xh, xl, wvh, wvl, bv, nullptr, nullptr, vbuf);

    dim3 g2(N_DIM / 128, B_DIM);
    attn_stats_kernel<<<g2, 256, 131072, stream>>>(qh, qlo, kh, klo, alpha, beta);

    vbsum_kernel<<<B_DIM * 32, 256, 0, stream>>>(vbuf, vbs);
    finalize_kernel<<<B_DIM * 32, 256, 0, stream>>>(vbuf, vbs, alpha, beta, (float*)d_out);
}
